// StandardAttentionWithRoPE_52201032515590
// MI455X (gfx1250) — compile-verified, matrix operand order checked by reading, not run on hardware
//
#include <hip/hip_runtime.h>
#include <hip/hip_bf16.h>

typedef __attribute__((ext_vector_type(16))) __bf16 v16bf;
typedef __attribute__((ext_vector_type(8)))  float  v8f;

#define EMBED   1024
#define NHEADS  16
#define HDIM    64
#define RHALF   32
#define BATCH   2
#define SEQ     2048

#define TM 128
#define TN 64
#define TK 32
#define LDP 40   // padded row (ushorts): 80B, multiple of 16B -> aligned b128 LDS ops

__device__ __forceinline__ unsigned short f2bf(float f) {
  unsigned int u = __float_as_uint(f);
  return (unsigned short)((u + 0x7FFFu + ((u >> 16) & 1u)) >> 16); // RNE
}

__device__ __forceinline__ v16bf make_frag(const unsigned short* p0, const unsigned short* p1) {
  union { v16bf v; uint4 u[2]; } t;
  t.u[0] = *(const uint4*)p0;   // elements 0-7  : K = ksel .. ksel+7
  t.u[1] = *(const uint4*)p1;   // elements 8-15 : K = 16+ksel .. 16+ksel+7
  return t.v;
}

__device__ __forceinline__ void cvt_store8(unsigned short* dst, float4 v) {
  union { __hip_bfloat162 h[2]; uint2 u; } pk;
  pk.h[0] = __float22bfloat162_rn(make_float2(v.x, v.y));
  pk.h[1] = __float22bfloat162_rn(make_float2(v.z, v.w));
  *(uint2*)dst = pk.u;   // ds_store_b64
}

// ---------------------------------------------------------------------------
// Generic GEMM: Y[M,N] = X[M,K] @ W[N,K]^T + bias,  fp32 in/out, bf16 WMMA math
// block = 256 threads (8 waves); tile 128x64; wave w -> rows [w*16, w*16+16)
// Staging: batched float4 loads (MLP) -> packed cvt -> b64 LDS stores.
// ---------------------------------------------------------------------------
__global__ __launch_bounds__(256) void gemm_bf16_kernel(
    const float* __restrict__ X, const float* __restrict__ W,
    const float* __restrict__ bias, float* __restrict__ Y,
    int M, int N, int K)
{
  __shared__ unsigned short As[TM][LDP];
  __shared__ unsigned short Bs[TN][LDP];

  const int t    = threadIdx.x;
  const int lane = t & 31;
  const int wave = t >> 5;
  const int m0   = blockIdx.y * TM;
  const int n0   = blockIdx.x * TN;
  const int ln   = lane & 15;
  const int ksel = (lane >> 4) * 8;

  // per-thread staging coordinates (8 threads per 32-wide row, float4 each)
  const int sr = t >> 3;          // 0..31
  const int sc = (t & 7) * 4;     // 0,4,...,28

  v8f acc[4] = {};

  for (int k0 = 0; k0 < K; k0 += TK) {
    // ---- batch all staging loads first (6 x global_load_b128 in flight)
    float4 xv0 = *(const float4*)&X[(long)(m0 + sr      ) * K + k0 + sc];
    float4 xv1 = *(const float4*)&X[(long)(m0 + sr + 32 ) * K + k0 + sc];
    float4 xv2 = *(const float4*)&X[(long)(m0 + sr + 64 ) * K + k0 + sc];
    float4 xv3 = *(const float4*)&X[(long)(m0 + sr + 96 ) * K + k0 + sc];
    float4 wv0 = *(const float4*)&W[(long)(n0 + sr      ) * K + k0 + sc];
    float4 wv1 = *(const float4*)&W[(long)(n0 + sr + 32 ) * K + k0 + sc];
    if (k0 + TK < K)   // warm L2/WGP$ for next K-slab
      __builtin_prefetch(&X[(long)(m0 + sr) * K + k0 + TK + sc], 0, 1);

    // ---- convert + store (b64 LDS stores)
    cvt_store8(&As[sr     ][sc], xv0);
    cvt_store8(&As[sr + 32][sc], xv1);
    cvt_store8(&As[sr + 64][sc], xv2);
    cvt_store8(&As[sr + 96][sc], xv3);
    cvt_store8(&Bs[sr     ][sc], wv0);
    cvt_store8(&Bs[sr + 32][sc], wv1);
    __syncthreads();

    v16bf a = make_frag(&As[wave * 16 + ln][ksel], &As[wave * 16 + ln][16 + ksel]);
    #pragma unroll
    for (int cg = 0; cg < 4; ++cg) {
      v16bf b = make_frag(&Bs[cg * 16 + ln][ksel], &Bs[cg * 16 + ln][16 + ksel]);
      acc[cg] = __builtin_amdgcn_wmma_f32_16x16x32_bf16(
          false, a, false, b, (short)0, acc[cg], false, false);
    }
    __syncthreads();
  }

  #pragma unroll
  for (int cg = 0; cg < 4; ++cg) {
    int col = n0 + cg * 16 + ln;
    float bv = bias[col];
    #pragma unroll
    for (int r = 0; r < 8; ++r) {
      int row = m0 + wave * 16 + r + 8 * (lane >> 4);   // 16x16 f32 C layout
      Y[(long)row * N + col] = acc[cg][r] + bv;
    }
  }
}

// ---------------------------------------------------------------------------
// RoPE on q,k + repack:
//   qh, kh : bf16 [B, H, N, 64]   (row-major, WMMA A / K^T-B friendly)
//   vt     : bf16 [B, H, 64, N]   (pre-transposed -> P@V B-fragments are
//                                  contiguous global_load_b128, no LDS stage)
// one thread per (b, n, h, pair j)
// ---------------------------------------------------------------------------
__global__ __launch_bounds__(256) void rope_pack_kernel(
    const float* __restrict__ q, const float* __restrict__ k, const float* __restrict__ v,
    const float* __restrict__ cosT, const float* __restrict__ sinT,
    unsigned short* __restrict__ qh, unsigned short* __restrict__ kh,
    unsigned short* __restrict__ vt)
{
  int tid = blockIdx.x * blockDim.x + threadIdx.x;   // B*N*H*RHALF = 2^21
  int j = tid & (RHALF - 1);
  int h = (tid >> 5) & (NHEADS - 1);
  int n = (tid >> 9) & (SEQ - 1);
  int b = tid >> 20;
  int bh = b * NHEADS + h;

  long src = ((long)(b * SEQ + n)) * EMBED + h * HDIM + 2 * j;
  // batch all loads before use
  float2 qv = *(const float2*)&q[src];
  float2 kv = *(const float2*)&k[src];
  float2 vv = *(const float2*)&v[src];
  float  c  = cosT[n * RHALF + j];
  float  s  = sinT[n * RHALF + j];

  long dst = (((long)bh) * SEQ + n) * HDIM + 2 * j;
  *(__hip_bfloat162*)&qh[dst] =
      __float22bfloat162_rn(make_float2(qv.x * c - qv.y * s, qv.x * s + qv.y * c));
  *(__hip_bfloat162*)&kh[dst] =
      __float22bfloat162_rn(make_float2(kv.x * c - kv.y * s, kv.x * s + kv.y * c));

  long tdst = ((long)bh * HDIM + 2 * j) * SEQ + n;
  vt[tdst]       = f2bf(vv.x);
  vt[tdst + SEQ] = f2bf(vv.y);
}

// ---------------------------------------------------------------------------
// Flash attention: one wave (32 threads) per 16-row Q tile per (b,h).
// Online softmax; P relayed C->A via tiny LDS buffer; K and V^T fragments
// loaded straight from global (L2-resident) as aligned b128s.
// ---------------------------------------------------------------------------
__global__ __launch_bounds__(32) void attn_kernel(
    const unsigned short* __restrict__ qh, const unsigned short* __restrict__ kh,
    const unsigned short* __restrict__ vt, float* __restrict__ out)
{
  __shared__ unsigned short Ps[16][LDP];     // P tile 16x32 bf16

  const int lane = threadIdx.x;
  const int bh = blockIdx.y;
  const int b = bh >> 4, h = bh & 15;
  const int q0 = blockIdx.x * 16;

  const unsigned short* qbase = qh + (long)bh * SEQ * HDIM;
  const unsigned short* kbase = kh + (long)bh * SEQ * HDIM;
  const unsigned short* vbase = vt + (long)bh * HDIM * SEQ;

  const int ln   = lane & 15;
  const int ksel = (lane >> 4) * 8;

  const unsigned short* qrow = qbase + (long)(q0 + ln) * HDIM;
  v16bf qa0 = make_frag(qrow + ksel,      qrow + 16 + ksel);   // head-dim 0..31
  v16bf qa1 = make_frag(qrow + 32 + ksel, qrow + 48 + ksel);   // head-dim 32..63

  float mrow[8], lrow[8];
  #pragma unroll
  for (int r = 0; r < 8; ++r) { mrow[r] = -3.0e38f; lrow[r] = 0.f; }
  v8f acc[4] = {};
  const float scale = 0.125f;   // 1/sqrt(64)

  for (int kb = 0; kb < SEQ; kb += 32) {
    // --- scores: two 16-column tiles (32 keys), K-dim 64 via 2 WMMAs each
    v8f sc[2];
    #pragma unroll
    for (int j = 0; j < 2; ++j) {
      const unsigned short* krow = kbase + (long)(kb + j * 16 + ln) * HDIM;
      v16bf kf0 = make_frag(krow + ksel,      krow + 16 + ksel);
      v16bf kf1 = make_frag(krow + 32 + ksel, krow + 48 + ksel);
      v8f s = {};
      s = __builtin_amdgcn_wmma_f32_16x16x32_bf16(false, qa0, false, kf0, (short)0, s, false, false);
      s = __builtin_amdgcn_wmma_f32_16x16x32_bf16(false, qa1, false, kf1, (short)0, s, false, false);
      sc[j] = s;
    }
    if (kb + 32 < SEQ)
      __builtin_prefetch(kbase + (long)(kb + 32 + ln) * HDIM, 0, 1);

    // --- online softmax (rows live in 16-lane halves of each C VGPR)
    #pragma unroll
    for (int r = 0; r < 8; ++r) {
      float s0 = sc[0][r] * scale;
      float s1 = sc[1][r] * scale;
      float cm = fmaxf(s0, s1);
      cm = fmaxf(cm, __shfl_xor(cm, 1, 32));
      cm = fmaxf(cm, __shfl_xor(cm, 2, 32));
      cm = fmaxf(cm, __shfl_xor(cm, 4, 32));
      cm = fmaxf(cm, __shfl_xor(cm, 8, 32));
      float mn = fmaxf(mrow[r], cm);
      float alpha = __expf(mrow[r] - mn);
      mrow[r] = mn;
      float p0 = __expf(s0 - mn);
      float p1 = __expf(s1 - mn);
      int row = r + 8 * (lane >> 4);
      Ps[row][ln]      = f2bf(p0);
      Ps[row][16 + ln] = f2bf(p1);
      float ps = p0 + p1;
      ps += __shfl_xor(ps, 1, 32);
      ps += __shfl_xor(ps, 2, 32);
      ps += __shfl_xor(ps, 4, 32);
      ps += __shfl_xor(ps, 8, 32);
      lrow[r] = lrow[r] * alpha + ps;
      #pragma unroll
      for (int cg = 0; cg < 4; ++cg) acc[cg][r] *= alpha;
    }
    __syncthreads();   // Ps visible (single wave -> just a dscnt wait)

    // --- O += P(16x32) @ V(32x64): A from Ps, B columns contiguous in vt
    v16bf pa = make_frag(&Ps[ln][ksel], &Ps[ln][16 + ksel]);
    #pragma unroll
    for (int cg = 0; cg < 4; ++cg) {
      const unsigned short* vcol = vbase + (long)(cg * 16 + ln) * SEQ + kb;
      v16bf vf = make_frag(vcol + ksel, vcol + 16 + ksel);
      acc[cg] = __builtin_amdgcn_wmma_f32_16x16x32_bf16(
          false, pa, false, vf, (short)0, acc[cg], false, false);
    }
    // next iteration's Ps stores are ordered after this iteration's Ps loads
    // by in-order LDS within the single wave; no extra barrier needed
  }

  // --- normalize and write [B, N, H*D] fp32 for output projection
  #pragma unroll
  for (int cg = 0; cg < 4; ++cg) {
    #pragma unroll
    for (int r = 0; r < 8; ++r) {
      int n = q0 + r + 8 * (lane >> 4);
      out[((long)(b * SEQ + n)) * EMBED + h * HDIM + cg * 16 + ln] = acc[cg][r] / lrow[r];
    }
  }
}

// ---------------------------------------------------------------------------
extern "C" void kernel_launch(void* const* d_in, const int* in_sizes, int n_in,
                              void* d_out, int out_size, void* d_ws, size_t ws_size,
                              hipStream_t stream)
{
  const float* query = (const float*)d_in[0];
  const float* key   = (const float*)d_in[1];
  const float* value = (const float*)d_in[2];
  const float* cosT  = (const float*)d_in[3];
  const float* sinT  = (const float*)d_in[4];
  const float* Wq = (const float*)d_in[5];
  const float* bq = (const float*)d_in[6];
  const float* Wk = (const float*)d_in[7];
  const float* bk = (const float*)d_in[8];
  const float* Wv = (const float*)d_in[9];
  const float* bv = (const float*)d_in[10];
  const float* Wo = (const float*)d_in[11];
  const float* bo = (const float*)d_in[12];
  float* outp = (float*)d_out;

  const long MT = (long)BATCH * SEQ;   // 4096 rows
  char* ws = (char*)d_ws;
  float* qf = (float*)ws;                    ws += MT * EMBED * sizeof(float);
  float* kf = (float*)ws;                    ws += MT * EMBED * sizeof(float);
  float* vf = (float*)ws;                    ws += MT * EMBED * sizeof(float);
  float* af = (float*)ws;                    ws += MT * EMBED * sizeof(float);
  unsigned short* qhb = (unsigned short*)ws; ws += MT * EMBED * sizeof(unsigned short);
  unsigned short* khb = (unsigned short*)ws; ws += MT * EMBED * sizeof(unsigned short);
  unsigned short* vtb = (unsigned short*)ws;

  dim3 gg(EMBED / TN, MT / TM);   // (16, 32)
  gemm_bf16_kernel<<<gg, 256, 0, stream>>>(query, Wq, bq, qf, (int)MT, EMBED, EMBED);
  gemm_bf16_kernel<<<gg, 256, 0, stream>>>(key,   Wk, bk, kf, (int)MT, EMBED, EMBED);
  gemm_bf16_kernel<<<gg, 256, 0, stream>>>(value, Wv, bv, vf, (int)MT, EMBED, EMBED);

  int total = BATCH * SEQ * NHEADS * RHALF;   // 2,097,152
  rope_pack_kernel<<<total / 256, 256, 0, stream>>>(qf, kf, vf, cosT, sinT, qhb, khb, vtb);

  attn_kernel<<<dim3(SEQ / 16, BATCH * NHEADS), 32, 0, stream>>>(qhb, khb, vtb, af);

  gemm_bf16_kernel<<<gg, 256, 0, stream>>>(af, Wo, bo, outp, (int)MT, EMBED, EMBED);
}
